// RIN_18897856102815
// MI455X (gfx1250) — matrix operand order repaired and checked by reading, not compiled, not run on hardware
//
#include <hip/hip_runtime.h>
#include <math.h>

typedef __bf16 bf16;
typedef __attribute__((ext_vector_type(16))) __bf16 v16bf;
typedef __attribute__((ext_vector_type(8)))  __bf16 v8bf;
typedef __attribute__((ext_vector_type(8)))  float  v8f;
typedef uint32_t u32x4 __attribute__((ext_vector_type(4)));
typedef uint32_t u32x8 __attribute__((ext_vector_type(8)));

#define DEV __device__ __forceinline__

constexpr int Bc = 1024, Tc = 128, Dc = 128, Hc = 256;
constexpr int BS = 32;                  // batch rows per WG in recurrent kernel
constexpr int MAIN_WGS = Bc / BS;       // 32

// ---- d_out layout (float elements) ----
constexpr size_t XIMP_OFF = 0;
constexpr size_t YOUT_OFF = (size_t)Bc * Tc * Dc;     // 16777216
constexpr size_t YSC_OFF  = YOUT_OFF + Bc;
constexpr size_t LOSS_OFF = YSC_OFF + Bc;
constexpr size_t XU_OFF   = LOSS_OFF + 1;
constexpr size_t XRS_OFF  = XU_OFF + (size_t)Bc * Tc * Dc;

// ---- workspace layout: bf16 weights then f32 scratch ----
constexpr size_t WDEC_E  = 0;                              // [256][128]
constexpr size_t WHIST_E = WDEC_E  + (size_t)Hc * Dc;      // [128][256]
constexpr size_t WV_E    = WHIST_E + (size_t)Dc * Hc;      // [128][128] diag-masked
constexpr size_t WR_E    = WV_E    + (size_t)Dc * Dc;      // [128][128] diag-masked
constexpr size_t WIH_E   = WR_E    + (size_t)Dc * Dc;      // [768][256]
constexpr size_t WHH_E   = WIH_E   + (size_t)3 * Hc * 2 * Dc; // [768][256]
constexpr size_t W_TOTAL_E = WHH_E + (size_t)3 * Hc * Hc;  // 491520 bf16
constexpr size_t NUM_BYTE  = W_TOTAL_E * 2;                // then: f32 num[128], den[128]

// ================= WMMA fragment helpers =================
// A operand: 16x32 bf16 tile, row-major in LDS with row stride `stride`.
// Lane l<16  : row l, K 0..15
// Lane l>=16 : row l-16, K 16..31
DEV v16bf frag_a(const bf16* tile, int stride, int k0) {
  int lane = threadIdx.x & 31;
  int r  = lane & 15;
  int kh = lane >> 4;                    // 0 or 1
  const bf16* p = tile + r * stride + k0 + kh * 16;
  v8bf lo = *(const v8bf*)(p);
  v8bf hi = *(const v8bf*)(p + 8);
  return __builtin_shufflevector(lo, hi, 0,1,2,3,4,5,6,7,8,9,10,11,12,13,14,15);
}

// B operand: 32x16 bf16 (K x N). W is the weight matrix [out][in] row-major,
// i.e. exactly B^T. Lane n<16: col n, K k0..k0+15 ; lane n>=16: K k0+16..k0+31.
DEV v16bf frag_b(const bf16* __restrict__ W, int ldw, int n0, int k0) {
  int lane = threadIdx.x & 31;
  int n  = lane & 15;
  int kh = lane >> 4;
  const bf16* p = W + (size_t)(n0 + n) * ldw + k0 + kh * 16;
  return *(const v16bf*)p;               // 32 contiguous bytes
}

DEV v8f wmma_bf16(v16bf a, v16bf b, v8f c) {
  return __builtin_amdgcn_wmma_f32_16x16x32_bf16(false, a, false, b, (short)0, c,
                                                 false, false);
}

DEV float sigmoidf(float x) { return 1.0f / (1.0f + expf(-x)); }

static __device__ const v8f VZERO = {0.f,0.f,0.f,0.f,0.f,0.f,0.f,0.f};

// ---- TDM: 2-D tile (rows x cols f32, row stride in elems) -> contiguous LDS ----
// D# per CDNA5 ISA 8.3/8.4. VADDR2/3 omitted (NULL) => 2-D tensor descriptor.
DEV void tdm_load_2d_f32(uint32_t lds_byte_off, const void* gptr,
                         uint32_t cols, uint32_t rows, uint32_t row_stride_elems) {
  uint64_t ga = (uint64_t)(uintptr_t)gptr;
  u32x4 g0;
  g0[0] = 0x1u;                                   // count=1, user mode
  g0[1] = lds_byte_off;                           // lds_addr
  g0[2] = (uint32_t)ga;                           // global_addr[31:0]
  g0[3] = (uint32_t)(ga >> 32) | (2u << 30);      // global_addr[56:32] | type=2
  u32x8 g1;
  g1[0] = 2u << 16;                               // data_size = 4 bytes
  g1[1] = (cols & 0xffffu) << 16;                 // tensor_dim0[15:0]
  g1[2] = (cols >> 16) | ((rows & 0xffffu) << 16);// tensor_dim0[31:16] | tensor_dim1[15:0]
  g1[3] = (rows >> 16) | ((cols & 0xffffu) << 16);// tensor_dim1[31:16] | tile_dim0
  g1[4] = rows & 0xffffu;                         // tile_dim1 | tile_dim2=0
  g1[5] = row_stride_elems;                       // tensor_dim0_stride[31:0]
  g1[6] = 0;                                      // stride0[47:32] | stride1[15:0]
  g1[7] = 0;                                      // stride1[47:16]
  asm volatile("tensor_load_to_lds %0, %1" :: "s"(g0), "s"(g1) : "memory");
}

// ================= kernel 1: weight convert (+mask) & zero num =================
__global__ __launch_bounds__(256) void prep_kernel(
    const float* __restrict__ Wd, const float* __restrict__ Wh,
    const float* __restrict__ Wv, const float* __restrict__ Wr,
    const float* __restrict__ Wih, const float* __restrict__ Whh,
    bf16* __restrict__ wsb, float* __restrict__ num) {
  size_t i = (size_t)blockIdx.x * 256 + threadIdx.x;
  if (i < 128) num[i] = 0.0f;
  if (i >= W_TOTAL_E) return;
  float v;
  if (i < WHIST_E)      v = Wd[i - WDEC_E];
  else if (i < WV_E)    v = Wh[i - WHIST_E];
  else if (i < WR_E) {  size_t l = i - WV_E;  int r = (int)(l >> 7), c = (int)(l & 127);
                        v = (r == c) ? 0.0f : Wv[l]; }
  else if (i < WIH_E) { size_t l = i - WR_E;  int r = (int)(l >> 7), c = (int)(l & 127);
                        v = (r == c) ? 0.0f : Wr[l]; }
  else if (i < WHH_E)   v = Wih[i - WIH_E];
  else                  v = Whh[i - WHH_E];
  wsb[i] = (bf16)v;
}

// ================= kernel 2: den[t] = sum(m[:,t,:]) =================
__global__ __launch_bounds__(256) void den_kernel(const float* __restrict__ m,
                                                  float* __restrict__ den) {
  int t = blockIdx.x;
  float s = 0.0f;
  for (int i = threadIdx.x; i < Bc * Dc; i += 256) {
    int b = i >> 7, c = i & 127;
    s += m[((size_t)b * Tc + t) * Dc + c];
  }
  for (int off = 16; off > 0; off >>= 1) s += __shfl_down(s, off);
  __shared__ float red[8];
  int wave = threadIdx.x >> 5;
  if ((threadIdx.x & 31) == 0) red[wave] = s;
  __syncthreads();
  if (threadIdx.x == 0) {
    float tot = 0.0f;
    for (int w = 0; w < 8; ++w) tot += red[w];
    den[t] = tot;
  }
}

// ================= kernel 3: xu = (m*x+(1-m)*x_hat) @ (Wv*off)^T + bv =================
__global__ __launch_bounds__(256) void xu_kernel(
    const float* __restrict__ x, const float* __restrict__ xh,
    const float* __restrict__ m, const bf16* __restrict__ Wv,
    const float* __restrict__ bv, float* __restrict__ out) {
  __shared__ __attribute__((aligned(16))) bf16 sh[128][128];
  int tid = threadIdx.x, wave = tid >> 5, lane = tid & 31;
  size_t R0 = (size_t)blockIdx.x * 128;   // flattened B*T row base
  for (int i = tid; i < 128 * 128; i += 256) {
    int r = i >> 7, c = i & 127;
    size_t gi = (R0 + r) * 128 + c;
    float mv = m[gi];
    sh[r][c] = (bf16)(mv * x[gi] + (1.0f - mv) * xh[gi]);
  }
  __syncthreads();
  for (int i = 0; i < 8; ++i) {
    int u = wave * 8 + i, mt = u >> 3, nt = u & 7;
    v8f acc = VZERO;
    #pragma unroll
    for (int k0 = 0; k0 < 128; k0 += 32)
      acc = wmma_bf16(frag_a(&sh[mt * 16][0], 128, k0), frag_b(Wv, 128, nt * 16, k0), acc);
    int n  = nt * 16 + (lane & 15);
    int mb = mt * 16 + (lane >> 4) * 8;
    float b = bv[n];
    #pragma unroll
    for (int j = 0; j < 8; ++j)
      out[XU_OFF + (R0 + mb + j) * 128 + n] = acc[j] + b;
  }
}

// ================= kernel 4: recurrent scan, 32 rows per WG =================
__global__ __launch_bounds__(256) void brits_main_kernel(
    const float* __restrict__ x,  const float* __restrict__ m,
    const float* __restrict__ d,
    const float* __restrict__ b_decay, const float* __restrict__ b_hist,
    const float* __restrict__ br, const float* __restrict__ conv_w,
    const float* __restrict__ conv_b,
    const float* __restrict__ b_ih, const float* __restrict__ b_hh,
    const float* __restrict__ W_out, const float* __restrict__ b_out,
    const bf16* __restrict__ Wdec, const bf16* __restrict__ Whist,
    const bf16* __restrict__ Wr,   const bf16* __restrict__ Wih,
    const bf16* __restrict__ Whh,
    float* __restrict__ out, float* __restrict__ num) {
  __shared__ __attribute__((aligned(16))) float sh_h[BS][Hc];      // f32 state (32KB)
  __shared__ __attribute__((aligned(16))) bf16  sh_hbf[BS][Hc];    // decayed h, bf16 (16KB)
  __shared__ __attribute__((aligned(16))) bf16  sh_a[BS][Hc];      // d_t / x_r / gin (16KB)
  __shared__ __attribute__((aligned(16))) float sh_stage[BS][Dc];  // TDM landing pad (16KB)

  int tid = threadIdx.x, wave = tid >> 5, lane = tid & 31;
  int row0 = blockIdx.x * BS;
  for (int i = tid; i < BS * Hc; i += 256) ((float*)sh_h)[i] = 0.0f;
  float cw0 = conv_w[0], cw1 = conv_w[1], cb = conv_b[0];
  uint32_t stage_lds = (uint32_t)(uintptr_t)&sh_stage[0][0];  // LDS byte offset
  __syncthreads();

  for (int t = 0; t < Tc; ++t) {
    // ---- phase 0: TDM: d_t tile [BS x Dc] f32 (row stride T*D) -> LDS, then bf16 ----
    if (wave == 0) {
      const float* gsrc = d + ((size_t)row0 * Tc + t) * Dc;
      tdm_load_2d_f32(stage_lds, gsrc, Dc, BS, Tc * Dc);
      __builtin_amdgcn_s_wait_tensorcnt(0);
    }
    __syncthreads();
    for (int i = tid; i < BS * Dc; i += 256) {
      int r = i >> 7, c = i & 127;
      sh_a[r][c] = (bf16)sh_stage[r][c];
    }
    __syncthreads();

    // ---- phase 1: gamma = exp(-relu(d@Wdec^T+b)); h *= gamma; also bf16 copy ----
    for (int i = 0; i < 4; ++i) {
      int u = wave + 8 * i, nt = u >> 1, mt = u & 1;
      v8f acc = VZERO;
      #pragma unroll
      for (int k0 = 0; k0 < Dc; k0 += 32)
        acc = wmma_bf16(frag_a(&sh_a[mt * 16][0], Hc, k0), frag_b(Wdec, Dc, nt * 16, k0), acc);
      int n  = nt * 16 + (lane & 15);
      int mb = mt * 16 + (lane >> 4) * 8;
      float bd = b_decay[n];
      #pragma unroll
      for (int j = 0; j < 8; ++j) {
        float g  = expf(-fmaxf(acc[j] + bd, 0.0f));
        float hv = sh_h[mb + j][n] * g;
        sh_h[mb + j][n]   = hv;
        sh_hbf[mb + j][n] = (bf16)hv;
      }
    }
    __syncthreads();

    // ---- phase 2: x_h = h@Whist^T+b; x_r = m*x+(1-m)*x_h -> sh_a (bf16) ----
    for (int i = 0; i < 2; ++i) {
      int u = wave * 2 + i, mt = u >> 3, nt = u & 7;
      v8f acc = VZERO;
      #pragma unroll
      for (int k0 = 0; k0 < Hc; k0 += 32)
        acc = wmma_bf16(frag_a(&sh_hbf[mt * 16][0], Hc, k0), frag_b(Whist, Hc, nt * 16, k0), acc);
      int n  = nt * 16 + (lane & 15);
      int mb = mt * 16 + (lane >> 4) * 8;
      float bh = b_hist[n];
      #pragma unroll
      for (int j = 0; j < 8; ++j) {
        size_t gi = ((size_t)(row0 + mb + j) * Tc + t) * Dc + n;
        float xv = x[gi], mv = m[gi];
        sh_a[mb + j][n] = (bf16)(mv * xv + (1.0f - mv) * (acc[j] + bh));
      }
    }
    __syncthreads();

    // ---- phase 3: xr = x_r@Wr_m^T+br; x_comb; loss; x_imp; build gin ----
    float lp = 0.0f;
    float ximp_s[2][8], mval_s[2][8];
    int n_s[2], mb_s[2];
    for (int i = 0; i < 2; ++i) {
      int u = wave * 2 + i, mt = u >> 3, nt = u & 7;
      v8f acc = VZERO;
      #pragma unroll
      for (int k0 = 0; k0 < Dc; k0 += 32)
        acc = wmma_bf16(frag_a(&sh_a[mt * 16][0], Hc, k0), frag_b(Wr, Dc, nt * 16, k0), acc);
      int n  = nt * 16 + (lane & 15);
      int mb = mt * 16 + (lane >> 4) * 8;
      n_s[i] = n; mb_s[i] = mb;
      float brv = br[n];
      #pragma unroll
      for (int j = 0; j < 8; ++j) {
        size_t gi = ((size_t)(row0 + mb + j) * Tc + t) * Dc + n;
        float xv = x[gi], mv = m[gi];
        float xrv = acc[j] + brv;
        out[XRS_OFF + gi] = xrv;
        float xc  = cw0 * out[XU_OFF + gi] + cw1 * xrv + cb;
        lp += fabsf(xv - xc) * mv;
        float xiv = mv * xv + (1.0f - mv) * xc;
        out[XIMP_OFF + gi] = xiv;
        ximp_s[i][j] = xiv; mval_s[i][j] = mv;
      }
    }
    for (int off = 16; off > 0; off >>= 1) lp += __shfl_down(lp, off);
    if (lane == 0) atomicAdd(&num[t], lp);
    __syncthreads();                      // all x_r reads done before gin overwrite
    for (int i = 0; i < 2; ++i) {
      #pragma unroll
      for (int j = 0; j < 8; ++j) {
        sh_a[mb_s[i] + j][n_s[i]]       = (bf16)ximp_s[i][j];   // gin[:, 0:128]  = x_imp
        sh_a[mb_s[i] + j][128 + n_s[i]] = (bf16)mval_s[i][j];   // gin[:,128:256] = m
      }
    }
    __syncthreads();

    // ---- phase 4: fused gi/gh GEMMs + GRU gates; update h in-place ----
    for (int i = 0; i < 4; ++i) {
      int u = wave + 8 * i, mt = u >> 4, ct = u & 15;
      int nr = ct * 16;
      v8f a0 = VZERO, a1 = VZERO, a2 = VZERO, a3 = VZERO, a4 = VZERO, a5 = VZERO;
      #pragma unroll
      for (int k0 = 0; k0 < 2 * Dc; k0 += 32) {
        v16bf ag = frag_a(&sh_a[mt * 16][0],   Hc, k0);   // gin
        v16bf ah = frag_a(&sh_hbf[mt * 16][0], Hc, k0);   // decayed h
        a0 = wmma_bf16(ag, frag_b(Wih, 2 * Dc, nr,           k0), a0); // i_r
        a1 = wmma_bf16(ag, frag_b(Wih, 2 * Dc, Hc + nr,      k0), a1); // i_z
        a2 = wmma_bf16(ag, frag_b(Wih, 2 * Dc, 2 * Hc + nr,  k0), a2); // i_n
        a3 = wmma_bf16(ah, frag_b(Whh, Hc,     nr,           k0), a3); // h_r
        a4 = wmma_bf16(ah, frag_b(Whh, Hc,     Hc + nr,      k0), a4); // h_z
        a5 = wmma_bf16(ah, frag_b(Whh, Hc,     2 * Hc + nr,  k0), a5); // h_n
      }
      int hc = nr + (lane & 15);
      int mb = mt * 16 + (lane >> 4) * 8;
      float bir = b_ih[hc], biz = b_ih[Hc + hc], bin_ = b_ih[2 * Hc + hc];
      float bhr = b_hh[hc], bhz = b_hh[Hc + hc], bhn  = b_hh[2 * Hc + hc];
      #pragma unroll
      for (int j = 0; j < 8; ++j) {
        float rg = sigmoidf(a0[j] + bir + a3[j] + bhr);
        float zg = sigmoidf(a1[j] + biz + a4[j] + bhz);
        float ng = tanhf(a2[j] + bin_ + rg * (a5[j] + bhn));
        float ho = sh_h[mb + j][hc];
        sh_h[mb + j][hc] = (1.0f - zg) * ng + zg * ho;
      }
    }
    __syncthreads();
  }

  // ---- output head: y = sigmoid(h_fin @ W_out^T + b_out) ----
  if (tid < BS) {
    float acc = b_out[0];
    for (int k = 0; k < Hc; ++k) acc += sh_h[tid][k] * W_out[k];
    out[YOUT_OFF + row0 + tid] = acc;
    out[YSC_OFF  + row0 + tid] = sigmoidf(acc);
  }
}

// ================= kernel 5: x_loss = sum_t num[t]/(den[t]+1e-5) =================
__global__ __launch_bounds__(128) void loss_kernel(const float* __restrict__ num,
                                                   const float* __restrict__ den,
                                                   float* __restrict__ out) {
  int tid = threadIdx.x;
  float v = num[tid] / (den[tid] + 1e-5f);
  for (int off = 16; off > 0; off >>= 1) v += __shfl_down(v, off);
  __shared__ float red[4];
  if ((tid & 31) == 0) red[tid >> 5] = v;
  __syncthreads();
  if (tid == 0) out[LOSS_OFF] = red[0] + red[1] + red[2] + red[3];
}

// ================= launch =================
extern "C" void kernel_launch(void* const* d_in, const int* in_sizes, int n_in,
                              void* d_out, int out_size, void* d_ws, size_t ws_size,
                              hipStream_t stream) {
  const float* x       = (const float*)d_in[0];
  const float* x_hat   = (const float*)d_in[1];
  // d_in[2] = u (unused by the reference forward pass)
  const float* m       = (const float*)d_in[3];
  const float* d       = (const float*)d_in[4];
  const float* W_decay = (const float*)d_in[5];
  const float* b_decay = (const float*)d_in[6];
  const float* W_hist  = (const float*)d_in[7];
  const float* b_hist  = (const float*)d_in[8];
  const float* Wv      = (const float*)d_in[9];
  const float* bv      = (const float*)d_in[10];
  const float* Wr      = (const float*)d_in[11];
  const float* br      = (const float*)d_in[12];
  const float* conv_w  = (const float*)d_in[13];
  const float* conv_b  = (const float*)d_in[14];
  const float* W_ih    = (const float*)d_in[15];
  const float* W_hh    = (const float*)d_in[16];
  const float* b_ih    = (const float*)d_in[17];
  const float* b_hh    = (const float*)d_in[18];
  const float* W_out   = (const float*)d_in[19];
  const float* b_out   = (const float*)d_in[20];

  bf16*  wsb  = (bf16*)d_ws;
  float* numf = (float*)((char*)d_ws + NUM_BYTE);
  float* denf = numf + 128;
  float* out  = (float*)d_out;

  int prep_blocks = (int)((W_TOTAL_E + 255) / 256);
  prep_kernel<<<prep_blocks, 256, 0, stream>>>(W_decay, W_hist, Wv, Wr, W_ih, W_hh,
                                               wsb, numf);
  den_kernel<<<Tc, 256, 0, stream>>>(m, denf);
  xu_kernel<<<(Bc * Tc) / 128, 256, 0, stream>>>(x, x_hat, m, wsb + WV_E, bv, out);
  brits_main_kernel<<<MAIN_WGS, 256, 0, stream>>>(
      x, m, d, b_decay, b_hist, br, conv_w, conv_b, b_ih, b_hh, W_out, b_out,
      wsb + WDEC_E, wsb + WHIST_E, wsb + WR_E, wsb + WIH_E, wsb + WHH_E, out, numf);
  loss_kernel<<<1, 128, 0, stream>>>(numf, denf, out);
}
